// ALISTA_40553081209409
// MI455X (gfx1250) — compile-verified, matrix operand order checked by reading, not run on hardware
//
#include <hip/hip_runtime.h>

typedef __attribute__((ext_vector_type(16))) __bf16 v16bf;
typedef __attribute__((ext_vector_type(8)))  float  v8f;

#define Mdim     256
#define Ndim     4096
#define Bdim     8192
#define KITER    16
#define RROWS    16
#define NTHREADS 256
#define XSTRIDE  4100   /* Ndim + 4: stride%64==4 -> conflict-free strided row reads */
#define BSTRIDE  260    /* Mdim + 4 */
#define AT_ELEMS (Ndim * Mdim)   /* 1,048,576 packed bf16 elems for A^T */
#define SMEM_BYTES ((size_t)(RROWS * XSTRIDE + RROWS * BSTRIDE) * sizeof(float)) /* 279,040 B */

static __device__ __forceinline__ v8f wmma_bf16(v16bf a, v16bf b, v8f c) {
  // D = A(16x32) * B(32x16) + C, f32 accumulate
  return __builtin_amdgcn_wmma_f32_16x16x32_bf16(false, a, false, b, (short)0, c,
                                                 false, false);
}

// Gather a 16x32 bf16 A-operand slice for this lane from an LDS-resident fp32
// row (rowbase = base of row (lane&15)). Assumed wave32 A layout: lanes 0-15
// row m=lane, k = {0..7 | 16..23}; lanes 16-31 same rows, k offset +8.
static __device__ __forceinline__ v16bf fetch_a_lds(const float* rowbase, int k0, int hh) {
  const float4* p0 = (const float4*)(rowbase + k0 + hh * 8);
  const float4* p1 = (const float4*)(rowbase + k0 + 16 + hh * 8);
  float4 a0 = p0[0];
  float4 a1 = p0[1];
  float4 a2 = p1[0];
  float4 a3 = p1[1];
  v16bf r;
  r[0]  = (__bf16)a0.x; r[1]  = (__bf16)a0.y; r[2]  = (__bf16)a0.z; r[3]  = (__bf16)a0.w;
  r[4]  = (__bf16)a1.x; r[5]  = (__bf16)a1.y; r[6]  = (__bf16)a1.z; r[7]  = (__bf16)a1.w;
  r[8]  = (__bf16)a2.x; r[9]  = (__bf16)a2.y; r[10] = (__bf16)a2.z; r[11] = (__bf16)a2.w;
  r[12] = (__bf16)a3.x; r[13] = (__bf16)a3.y; r[14] = (__bf16)a3.z; r[15] = (__bf16)a3.w;
  return r;
}

// ---------------------------------------------------------------------------
// One-time pack: A^T (4096x256) and W (256x4096) -> bf16 tiles in the per-lane
// WMMA B-operand layout. Tile (kt,nt) is 32x16; storage:
//   pack[((kt*NT + nt)*32 + lane)*16 + j], element (k = kt*32 + kmap(j,half),
//   n = nt*16 + (lane&15)), kmap(j,h) = (j&7) + ((j>>3)<<4) + (h<<3).
// ---------------------------------------------------------------------------
__global__ void alista_pack(const float* __restrict__ A, const float* __restrict__ W,
                            __bf16* __restrict__ packAt, __bf16* __restrict__ packW) {
  const int idx   = blockIdx.x * NTHREADS + threadIdx.x;
  const int local = idx & (AT_ELEMS - 1);
  const int j     = local & 15;
  const int L     = (local >> 4) & 31;
  const int tile  = local >> 9;
  const int hh    = L >> 4;
  const int nl    = L & 15;
  const int kk    = (j & 7) + ((j >> 3) << 4) + (hh << 3);
  if (idx < AT_ELEMS) {
    // A^T: K over Ndim (128 k-tiles), N over Mdim (16 n-tiles)
    const int kt = tile >> 4, nt = tile & 15;
    const int k = kt * 32 + kk, n = nt * 16 + nl;
    packAt[local] = (__bf16)A[(size_t)n * Ndim + k]; // A^T[k][n] = A[n][k]
  } else {
    // W: K over Mdim (8 k-tiles), N over Ndim (256 n-tiles)
    const int kt = tile >> 8, nt = tile & 255;
    const int k = kt * 32 + kk, n = nt * 16 + nl;
    packW[local] = (__bf16)W[(size_t)k * Ndim + n];
  }
}

// ---------------------------------------------------------------------------
// Fused ALISTA: 16 rows per WG stay resident in LDS for all 16 iterations.
// ---------------------------------------------------------------------------
__global__ __launch_bounds__(NTHREADS, 1) void alista_main(
    const float* __restrict__ y, const float* __restrict__ gamma,
    const float* __restrict__ theta, const __bf16* __restrict__ packAt,
    const __bf16* __restrict__ packW, float* __restrict__ out) {
  extern __shared__ char smem[];
  float* xf  = (float*)smem;           // [RROWS][XSTRIDE] fp32 master copy of x
  float* bsm = xf + RROWS * XSTRIDE;   // [RROWS][BSTRIDE] residual b = x A^T - y

  const int tid  = threadIdx.x;
  const int lane = tid & 31;
  const int wv   = tid >> 5;
  const int hh   = lane >> 4;
  const int ln   = lane & 15;
  const int row0 = blockIdx.x * RROWS;

  if (blockIdx.x == 0 && tid < 32)
    out[(size_t)Bdim * Ndim + tid] = 0.0f;  // the two (K,1) zero outputs

  { // x = 0 (zero the whole padded region)
    float4* x4 = (float4*)xf;
    float4  z; z.x = z.y = z.z = z.w = 0.0f;
    for (int i = tid; i < RROWS * XSTRIDE / 4; i += NTHREADS) x4[i] = z;
  }
  __syncthreads();

  const v16bf* pAt = (const v16bf*)packAt;  // one v16bf = one lane's tile slice
  const v16bf* pW  = (const v16bf*)packW;

  for (int it = 0; it < KITER; ++it) {
    const float g  = gamma[it];
    const float th = theta[it];

    // ---- Phase 1: b = x @ A^T - y  (16 x 256), K = 4096 ----
    {
      const int nt0 = wv * 2, nt1 = wv * 2 + 1;
      v8f acc0 = {}; v8f acc1 = {};
      const float* rb = xf + ln * XSTRIDE;
      for (int kt = 0; kt < Ndim / 32; ++kt) {
        v16bf a  = fetch_a_lds(rb, kt * 32, hh);
        v16bf b0 = pAt[(size_t)((kt * (Mdim / 16) + nt0) * 32 + lane)];
        v16bf b1 = pAt[(size_t)((kt * (Mdim / 16) + nt1) * 32 + lane)];
        acc0 = wmma_bf16(a, b0, acc0);
        acc1 = wmma_bf16(a, b1, acc1);
      }
#pragma unroll
      for (int r = 0; r < 8; ++r) {
        const int row = r + hh * 8;
        const float y0 = y[(size_t)(row0 + row) * Mdim + nt0 * 16 + ln];
        const float y1 = y[(size_t)(row0 + row) * Mdim + nt1 * 16 + ln];
        bsm[row * BSTRIDE + nt0 * 16 + ln] = acc0[r] - y0;
        bsm[row * BSTRIDE + nt1 * 16 + ln] = acc1[r] - y1;
      }
    }
    __syncthreads();

    // ---- Phase 2: x := x - g * (b @ W)  (16 x 4096), K = 256 ----
    {
      v16bf Ab[Mdim / 32];           // hoist all 8 K-step A-operands (64 VGPRs)
      const float* rb = bsm + ln * BSTRIDE;
#pragma unroll
      for (int kt = 0; kt < Mdim / 32; ++kt) Ab[kt] = fetch_a_lds(rb, kt * 32, hh);
      for (int t = 0; t < (Ndim / 16) / 8; ++t) {
        const int nt = t * 8 + wv;   // interleave waves across n for L2 locality
        v8f acca = {}; v8f accb = {};  // two chains -> break D->C RAW serialization
#pragma unroll
        for (int kt = 0; kt < Mdim / 32; kt += 2) {
          v16bf b0 = pW[(size_t)(((kt    ) * (Ndim / 16) + nt) * 32 + lane)];
          v16bf b1 = pW[(size_t)(((kt + 1) * (Ndim / 16) + nt) * 32 + lane)];
          acca = wmma_bf16(Ab[kt],     b0, acca);
          accb = wmma_bf16(Ab[kt + 1], b1, accb);
        }
#pragma unroll
        for (int r = 0; r < 8; ++r) {
          const int row = r + hh * 8;
          const int col = nt * 16 + ln;
          xf[row * XSTRIDE + col] -= g * (acca[r] + accb[r]);
        }
      }
    }
    __syncthreads();

    // ---- Phase 3: per-row top-128 soft threshold (wave owns 2 rows) ----
    for (int rsel = 0; rsel < 2; ++rsel) {
      const int row = wv * 2 + rsel;
      float* rp = xf + row * XSTRIDE;
      unsigned kp[64];               // 128 abs-keys (15-bit), packed pairs
#pragma unroll
      for (int c = 0; c < 32; ++c) {
        const uint4 vv = *(const uint4*)(rp + c * 128 + lane * 4);
        const unsigned k0 = (vv.x & 0x7fffffffu) >> 16;
        const unsigned k1 = (vv.y & 0x7fffffffu) >> 16;
        const unsigned k2 = (vv.z & 0x7fffffffu) >> 16;
        const unsigned k3 = (vv.w & 0x7fffffffu) >> 16;
        kp[c * 2]     = k0 | (k1 << 16);
        kp[c * 2 + 1] = k2 | (k3 << 16);
      }
      // radix binary search: largest T with count(key >= T) >= 128
      unsigned cur = 0;
      for (int bit = 14; bit >= 0; --bit) {
        const unsigned T = cur | (1u << bit);
        int cnt = 0;
#pragma unroll
        for (int i = 0; i < 64; ++i) {
          cnt += (int)((kp[i] & 0xffffu) >= T);
          cnt += (int)((kp[i] >> 16) >= T);
        }
        cnt += __shfl_xor(cnt, 16, 32);
        cnt += __shfl_xor(cnt, 8, 32);
        cnt += __shfl_xor(cnt, 4, 32);
        cnt += __shfl_xor(cnt, 2, 32);
        cnt += __shfl_xor(cnt, 1, 32);
        if (cnt >= 128) cur = T;
      }
      // apply: keep if key > thr, else soft-threshold by theta
#pragma unroll
      for (int c = 0; c < 32; ++c) {
        float4 v = *(float4*)(rp + c * 128 + lane * 4);
        float* e = (float*)&v;
#pragma unroll
        for (int q = 0; q < 4; ++q) {
          const float xv = e[q];
          const unsigned key = (__float_as_uint(xv) & 0x7fffffffu) >> 16;
          if (key <= cur) {
            float s = fabsf(xv) - th;
            s = s > 0.0f ? s : 0.0f;
            e[q] = copysignf(s, xv);
          }
        }
        *(float4*)(rp + c * 128 + lane * 4) = v;
      }
    }
    __syncthreads();
  }

  // ---- write back this WG's 16 rows (strided LDS -> contiguous global) ----
  {
    float4* dst = (float4*)(out + (size_t)row0 * Ndim);
    for (int i = tid; i < RROWS * Ndim / 4; i += NTHREADS) {
      const int row = i >> 10;            // Ndim/4 = 1024 float4 per row
      const int c4  = i & 1023;
      dst[(size_t)row * (Ndim / 4) + c4] = *(const float4*)(xf + row * XSTRIDE + c4 * 4);
    }
  }
}

extern "C" void kernel_launch(void* const* d_in, const int* in_sizes, int n_in,
                              void* d_out, int out_size, void* d_ws, size_t ws_size,
                              hipStream_t stream) {
  const float* y     = (const float*)d_in[0];
  const float* A     = (const float*)d_in[1];
  const float* W     = (const float*)d_in[2];
  const float* gamma = (const float*)d_in[3];
  const float* theta = (const float*)d_in[4];
  float* out = (float*)d_out;

  __bf16* packAt = (__bf16*)d_ws;            // 2 MB
  __bf16* packW  = packAt + AT_ELEMS;        // 2 MB
  (void)in_sizes; (void)n_in; (void)out_size; (void)ws_size;

  hipFuncSetAttribute((const void*)alista_main,
                      hipFuncAttributeMaxDynamicSharedMemorySize, (int)SMEM_BYTES);

  alista_pack<<<(2 * AT_ELEMS) / NTHREADS, NTHREADS, 0, stream>>>(A, W, packAt, packW);
  alista_main<<<Bdim / RROWS, NTHREADS, SMEM_BYTES, stream>>>(y, gamma, theta,
                                                              packAt, packW, out);
}
